// AbstractSNClustering_79491254714856
// MI455X (gfx1250) — matrix operand order repaired and checked by reading, not compiled, not run on hardware
//
#include <hip/hip_runtime.h>
#include <hip/hip_bf16.h>

typedef __attribute__((ext_vector_type(16))) _Float16 v16h;
typedef __attribute__((ext_vector_type(8)))  float    v8f;

#define NROWS 1048576
#define D     64
#define KCL   16
#define H1S   128
#define H2S   64
#define PROJ  6
#define EHS   32
#define WAVES 4   // waves per block; 16 rows per wave

// ---------------------------------------------------------------------------
// Fused kernel: cluster assignment (WMMA) + MLP layers 0/1 (WMMA) + expert MLP
// + blend. Writes unmasked prediction, cid, and accumulates global histogram.
// ---------------------------------------------------------------------------
__global__ void __launch_bounds__(32 * WAVES)
fused_kernel(const float* __restrict__ x, const float* __restrict__ s_in,
             const float* __restrict__ np_in, const float* __restrict__ centers,
             const float* __restrict__ W0, const float* __restrict__ b0,
             const float* __restrict__ W1, const float* __restrict__ b1,
             const float* __restrict__ W2, const float* __restrict__ b2,
             const float* __restrict__ EW0, const float* __restrict__ Eb0,
             const float* __restrict__ EW1, const float* __restrict__ Eb1,
             int* __restrict__ cids, int* __restrict__ counts,
             float* __restrict__ out) {
    // Transposed f16 weights: [n][k] so one lane's 16 consecutive K are contiguous.
    __shared__ __align__(32) _Float16 sW0T[H1S][D];    // 16 KB
    __shared__ __align__(32) _Float16 sW1T[H2S][H1S];  // 16 KB
    __shared__ __align__(32) _Float16 sCentH[KCL][D];  // -2*centers, 2 KB ([n][k] == row-major)
    __shared__ float sccv[KCL];                        // ||center_n||^2
    __shared__ float sW0r64[H1S];                      // W0 row 64 (naive_pred weight)
    __shared__ float sW2[H2S][PROJ];
    __shared__ float sb0[H1S], sb1[H2S], sb2[8];
    __shared__ float sEb0[KCL][EHS];
    __shared__ float sEb1[KCL][PROJ];
    __shared__ int   hist[KCL];
    // Per-wave stage: score tile f32 (1KB) -> h0 f16 (4KB) -> h1 f32 (4KB)
    __shared__ __align__(32) unsigned char sStage[WAVES][16 * H1S * 2];

    const int tid = threadIdx.x;
    // --- cooperative staging ---
    for (int i = tid; i < D * H1S; i += 32 * WAVES) {
        int k = i / H1S, n = i % H1S;
        sW0T[n][k] = (_Float16)W0[i];
    }
    for (int i = tid; i < H1S * H2S; i += 32 * WAVES) {
        int k = i / H2S, n = i % H2S;
        sW1T[n][k] = (_Float16)W1[i];
    }
    for (int i = tid; i < KCL * D; i += 32 * WAVES)
        sCentH[i / D][i % D] = (_Float16)(-2.f * centers[i]);
    if (tid < KCL) {
        float s = 0.f;
        for (int k = 0; k < D; ++k) { float c = centers[tid * D + k]; s += c * c; }
        sccv[tid] = s;
        hist[tid] = 0;
    }
    for (int i = tid; i < H1S; i += 32 * WAVES) { sW0r64[i] = W0[D * H1S + i]; sb0[i] = b0[i]; }
    for (int i = tid; i < H2S; i += 32 * WAVES) sb1[i] = b1[i];
    for (int i = tid; i < H2S * PROJ; i += 32 * WAVES) sW2[i / PROJ][i % PROJ] = W2[i];
    for (int i = tid; i < PROJ; i += 32 * WAVES) sb2[i] = b2[i];
    for (int i = tid; i < KCL * EHS; i += 32 * WAVES) sEb0[i / EHS][i % EHS] = Eb0[i];
    for (int i = tid; i < KCL * PROJ; i += 32 * WAVES) sEb1[i / PROJ][i % PROJ] = Eb1[i];
    __syncthreads();

    const int wv   = tid >> 5;
    const int lane = tid & 31;
    const int g    = lane >> 4;   // lane group
    const int m    = lane & 15;   // row within tile (A layout) / col N (B,C layout)
    const int row0 = (blockIdx.x * WAVES + wv) * 16;

    float*    stScore = (float*)sStage[wv];      // 16x16 f32 score tile
    _Float16* stH0    = (_Float16*)sStage[wv];   // 16x128 f16
    float*    stH1    = (float*)sStage[wv];      // 16x64 f32

    // --- build A fragments (16x32 f16, K-chunks 0 and 1) from global x ---
    // A layout: lane (g,m) holds row m; half-slot t -> k = 16*g + t
    v16h a0, a1;
#pragma unroll
    for (int t = 0; t < 16; ++t) {
        int i  = t >> 1, h = t & 1;
        int kb = t + 16 * g;
        a0[t] = (_Float16)x[(row0 + m) * D + kb];
        a1[t] = (_Float16)x[(row0 + m) * D + 32 + kb];
    }
    // naive_pred for rows this lane touches in C layout (M = r + 8*g)
    float npv[8];
#pragma unroll
    for (int r = 0; r < 8; ++r) npv[r] = np_in[row0 + r + 8 * g];

    // --- clustering: score = ||c_n||^2 - 2 * x.c_n  (one WMMA tile, 2 K-chunks) ---
    {
        v8f acc;
#pragma unroll
        for (int r = 0; r < 8; ++r) acc[r] = sccv[m];
        v16h bfr = *(const v16h*)&sCentH[m][16 * g];
        acc = __builtin_amdgcn_wmma_f32_16x16x32_f16(false, a0, false, bfr,
                                                     (short)0, acc, false, false);
        bfr = *(const v16h*)&sCentH[m][32 + 16 * g];
        acc = __builtin_amdgcn_wmma_f32_16x16x32_f16(false, a1, false, bfr,
                                                     (short)0, acc, false, false);
#pragma unroll
        for (int r = 0; r < 8; ++r) stScore[(r + 8 * g) * KCL + m] = acc[r];
    }
    int mycid = 0;
    if (lane < 16) {  // per-row argmin over the 16 cluster scores
        float best = 3.4e38f;
#pragma unroll
        for (int n = 0; n < KCL; ++n) {
            float v = stScore[lane * KCL + n];
            if (v < best) { best = v; mycid = n; }
        }
        cids[row0 + lane] = mycid;
        atomicAdd(&hist[mycid], 1);
    }

    // --- layer 0: 8 N-tiles x 2 K-chunks of WMMA (stage reused, same-wave DS order) ---
    for (int j = 0; j < 8; ++j) {
        const int n = 16 * j + m;
        v8f acc;
#pragma unroll
        for (int r = 0; r < 8; ++r) acc[r] = sb0[n] + npv[r] * sW0r64[n];
        v16h bfr = *(const v16h*)&sW0T[n][16 * g];
        acc = __builtin_amdgcn_wmma_f32_16x16x32_f16(false, a0, false, bfr,
                                                     (short)0, acc, false, false);
        bfr = *(const v16h*)&sW0T[n][32 + 16 * g];
        acc = __builtin_amdgcn_wmma_f32_16x16x32_f16(false, a1, false, bfr,
                                                     (short)0, acc, false, false);
#pragma unroll
        for (int r = 0; r < 8; ++r)
            stH0[(r + 8 * g) * H1S + n] = (_Float16)fmaxf(acc[r], 0.f);
    }

    // --- layer 1: rebuild A fragments (4 K-chunks of 32) from staged h0 ---
    v16h A1[4];
#pragma unroll
    for (int c = 0; c < 4; ++c)
#pragma unroll
        for (int t = 0; t < 16; ++t) {
            int i  = t >> 1, h = t & 1;
            int kb = 32 * c + t + 16 * g;
            A1[c][t] = stH0[m * H1S + kb];
        }
    for (int j = 0; j < 4; ++j) {
        const int n = 16 * j + m;
        v8f acc;
#pragma unroll
        for (int r = 0; r < 8; ++r) acc[r] = sb1[n];
#pragma unroll
        for (int c = 0; c < 4; ++c) {
            v16h bfr = *(const v16h*)&sW1T[n][32 * c + 16 * g];
            acc = __builtin_amdgcn_wmma_f32_16x16x32_f16(false, A1[c], false, bfr,
                                                         (short)0, acc, false, false);
        }
#pragma unroll
        for (int r = 0; r < 8; ++r)
            stH1[(r + 8 * g) * H2S + n] = fmaxf(acc[r], 0.f);
    }

    // --- scalar tail: lanes 0..15 each own one row (unmasked prediction) ---
    if (lane < 16) {
        const int   row = row0 + lane;
        const float np  = np_in[row];
        float hr[H2S];
#pragma unroll
        for (int q = 0; q < H2S; ++q) hr[q] = stH1[lane * H2S + q];

        float cp[PROJ];
#pragma unroll
        for (int p = 0; p < PROJ; ++p) {
            float s = sb2[p] + np;
            for (int q = 0; q < H2S; ++q) s += hr[q] * sW2[q][p];
            cp[p] = s;
        }

        const float* ew0 = EW0 + mycid * PROJ * EHS;  // [6][32]
        const float* ew1 = EW1 + mycid * EHS * PROJ;  // [32][6]

        float eh[EHS];
#pragma unroll 4
        for (int o = 0; o < EHS; ++o) {
            float s = sEb0[mycid][o];
#pragma unroll
            for (int p = 0; p < PROJ; ++p) s += cp[p] * ew0[p * EHS + o];
            eh[o] = fmaxf(s, 0.f);
        }
        float cp2[PROJ];
#pragma unroll
        for (int p = 0; p < PROJ; ++p) {
            float s = sEb1[mycid][p] + cp[p];
            for (int o = 0; o < EHS; ++o) s += eh[o] * ew1[o * PROJ + p];
            cp2[p] = s;
        }

        const float sv    = s_in[row];
        const float p_lin = cp2[0] - fabsf(cp2[1]) * sv + np;
        const float p_log = cp2[2] - fabsf(cp2[3]) * log10f(fabsf(sv) + 1.f) + np;
        const float a4 = fabsf(cp2[4]), a5 = fabsf(cp2[5]);
        const float den  = fmaxf(a4 + a5, 1e-12f);
        out[row] = (p_lin * a4 + p_log * a5) / den;
    }

    __syncthreads();
    if (tid < KCL) atomicAdd(&counts[tid], hist[tid]);
}

// ---------------------------------------------------------------------------
// Finalize: patch rows whose cluster has < 2 members back to naive_pred.
// ---------------------------------------------------------------------------
__global__ void __launch_bounds__(256)
finalize_kernel(const int* __restrict__ cids, const int* __restrict__ counts,
                const float* __restrict__ np_in, float* __restrict__ out) {
    const int row = blockIdx.x * 256 + threadIdx.x;
    const int c   = cids[row];
    if (counts[c] < 2) out[row] = np_in[row];
}

// ---------------------------------------------------------------------------
extern "C" void kernel_launch(void* const* d_in, const int* in_sizes, int n_in,
                              void* d_out, int out_size, void* d_ws, size_t ws_size,
                              hipStream_t stream) {
    const float* x       = (const float*)d_in[0];
    const float* s_in    = (const float*)d_in[1];
    const float* np_in   = (const float*)d_in[2];
    const float* centers = (const float*)d_in[3];
    const float* W0  = (const float*)d_in[4];
    const float* b0  = (const float*)d_in[5];
    const float* W1  = (const float*)d_in[6];
    const float* b1  = (const float*)d_in[7];
    const float* W2  = (const float*)d_in[8];
    const float* b2  = (const float*)d_in[9];
    const float* EW0 = (const float*)d_in[10];
    const float* Eb0 = (const float*)d_in[11];
    const float* EW1 = (const float*)d_in[12];
    const float* Eb1 = (const float*)d_in[13];
    float* out = (float*)d_out;

    int* counts = (int*)d_ws;       // 16 ints
    int* cids   = counts + 16;      // N ints

    hipMemsetAsync(d_ws, 0, KCL * sizeof(int), stream);
    fused_kernel<<<NROWS / (16 * WAVES), 32 * WAVES, 0, stream>>>(
        x, s_in, np_in, centers, W0, b0, W1, b1, W2, b2, EW0, Eb0, EW1, Eb1,
        cids, counts, out);
    finalize_kernel<<<NROWS / 256, 256, 0, stream>>>(cids, counts, np_in, out);
}
